// MaskRCNN_72103910965674
// MI455X (gfx1250) — matrix operand order repaired and checked by reading, not compiled, not run on hardware
//
#include <hip/hip_runtime.h>
#include <hip/hip_bf16.h>

// ---------------------------------------------------------------------------
// Mask R-CNN head for MI455X (gfx1250): all GEMM-shaped work runs through
// v_wmma_f32_16x16x32_bf16 with fp32 accumulation; intermediates kept in bf16
// to halve HBM traffic (23.3 TB/s).  Wave tile = 16x64 (1 A-frag x 4 B-frags),
// block = 256 threads = 8 waves = 128x64 tile.
// ---------------------------------------------------------------------------

typedef __attribute__((ext_vector_type(16))) __bf16 v16bf;
typedef __attribute__((ext_vector_type(8)))  __bf16 v8bf;
typedef __attribute__((ext_vector_type(8)))  float  v8f;

#define NUMC      81
#define NPROP     200
#define FH        256
#define FW        256
#define FC_CH     256
#define K_FC1     12544          // 256*7*7
#define N_FC      1024
#define SCALE     0.0625f
#define IMGSZ     1024.0f
#define P_DC1     39200          // 200*14*14
#define P_DC2     156800         // 200*28*28
#define MASK_P    3136           // 56*56

// ----------------------------- helpers -------------------------------------

__device__ __forceinline__ __bf16 f2b(float f) {
  union { float f; unsigned u; } v; v.f = f;
  unsigned r = v.u + 0x7FFFu + ((v.u >> 16) & 1u);   // RTNE
  union { unsigned short s; __bf16 b; } o; o.s = (unsigned short)(r >> 16);
  return o.b;
}

// Core wave tile: D[16 x 64] += A[16 x K] * B[K x 64], A row-major [M,K],
// Bt row-major transposed weights [N,K].  Fragments use contiguous k order
// (k = 16*(lane/16) + s) in both A and B, not the ISA table order.
__device__ __forceinline__ void gemm_tile(const __bf16* __restrict__ A,
                                          const __bf16* __restrict__ Bt,
                                          int lda, int ldb, int K,
                                          int row, bool row_ok, int col0,
                                          int lane, v8f acc[4]) {
  const int kbA = (lane & 16) ? 16 : 0;   // A: lanes 16-31 hold K 16-31
  const int kbB = (lane & 16) ? 16 : 0;   // B: lanes 16-31 hold K 16-31
  const int nn  = lane & 15;
  const __bf16* arow = A + (long)row * lda;
  const __bf16* brow = Bt + (long)(col0 + nn) * ldb + kbB;
  for (int k0 = 0; k0 < K; k0 += 32) {
    v16bf a;
    if (row_ok) {
      v8bf lo = *(const v8bf*)(arow + k0 + kbA);        // 16B aligned
      v8bf hi = *(const v8bf*)(arow + k0 + 8 + kbA);
#pragma unroll
      for (int i = 0; i < 8; ++i) { a[i] = lo[i]; a[i + 8] = hi[i]; }
    } else {
#pragma unroll
      for (int i = 0; i < 16; ++i) a[i] = (__bf16)0.0f;
    }
#pragma unroll
    for (int j = 0; j < 4; ++j) {
      v16bf b = *(const v16bf*)(brow + (long)(j * 16) * ldb + k0);  // 32B
      acc[j] = __builtin_amdgcn_wmma_f32_16x16x32_bf16(
          false, a, false, b, (short)0, acc[j], false, false);
    }
  }
}

// ----------------------- layout / conversion kernels -----------------------

// feature CHW -> HWC (coalesced channel gathers for 14x14 ROIAlign)
__global__ void feat_to_hwc(const float* __restrict__ f, float* __restrict__ o) {
  __shared__ float t[16][17];
  int pix0 = blockIdx.x * 16;            // 65536/16 tiles
  int c0   = blockIdx.y * 16;            // 256/16 tiles
  int tx = threadIdx.x, ty = threadIdx.y;
  t[ty][tx] = f[(long)(c0 + ty) * (FH * FW) + pix0 + tx];
  __syncthreads();
  o[(long)(pix0 + ty) * FC_CH + c0 + tx] = t[tx][ty];
}

// fp32 [K,N] -> bf16 transposed [Npad,K] (zero padded rows)
__global__ void w_transpose_bf16(const float* __restrict__ src,
                                 __bf16* __restrict__ dst,
                                 int K, int N, long total) {
  long idx = (long)blockIdx.x * 256 + threadIdx.x;
  if (idx >= total) return;
  int n = (int)(idx / K), k = (int)(idx % K);
  float v = (n < N) ? src[(long)k * N + n] : 0.0f;
  dst[idx] = f2b(v);
}

// wt (c,o,a,b) -> 4 matrices [ab][o][c] bf16 (B-transposed for deconv GEMMs)
__global__ void wt_convert(const float* __restrict__ wt, __bf16* __restrict__ dst) {
  int idx = blockIdx.x * 256 + threadIdx.x;   // 4*256*256
  int ab = idx >> 16, r = idx & 65535, o = r >> 8, c = r & 255;
  int a = ab >> 1, b = ab & 1;
  dst[idx] = f2b(wt[((c * 256 + o) * 2 + a) * 2 + b]);
}

// wm [81,256] -> bf16 padded [128,256]   (A operand of mask GEMM)
__global__ void wm_convert(const float* __restrict__ wm, __bf16* __restrict__ dst) {
  int idx = blockIdx.x * 256 + threadIdx.x;   // 128*256
  int o = idx >> 8, c = idx & 255;
  dst[idx] = f2b(o < NUMC ? wm[o * 256 + c] : 0.0f);
}

// ------------------------------ ROI align ----------------------------------

__device__ __forceinline__ float bilin_chw(const float* __restrict__ f,
                                           float Y, float X) {
  float y0 = floorf(Y), x0 = floorf(X);
  float ly = Y - y0, lx = X - x0;
  int y0i = (int)fminf(fmaxf(y0,        0.0f), (float)(FH - 1));
  int y1i = (int)fminf(fmaxf(y0 + 1.0f, 0.0f), (float)(FH - 1));
  int x0i = (int)fminf(fmaxf(x0,        0.0f), (float)(FW - 1));
  int x1i = (int)fminf(fmaxf(x0 + 1.0f, 0.0f), (float)(FW - 1));
  float v00 = f[y0i * FW + x0i], v01 = f[y0i * FW + x1i];
  float v10 = f[y1i * FW + x0i], v11 = f[y1i * FW + x1i];
  return v00 * (1 - ly) * (1 - lx) + v01 * (1 - ly) * lx +
         v10 * ly * (1 - lx) + v11 * ly * lx;
}

__device__ __forceinline__ float bilin_hwc(const float* __restrict__ f,
                                           float Y, float X, int c) {
  float y0 = floorf(Y), x0 = floorf(X);
  float ly = Y - y0, lx = X - x0;
  int y0i = (int)fminf(fmaxf(y0,        0.0f), (float)(FH - 1));
  int y1i = (int)fminf(fmaxf(y0 + 1.0f, 0.0f), (float)(FH - 1));
  int x0i = (int)fminf(fmaxf(x0,        0.0f), (float)(FW - 1));
  int x1i = (int)fminf(fmaxf(x0 + 1.0f, 0.0f), (float)(FW - 1));
  float v00 = f[(y0i * FW + x0i) * FC_CH + c], v01 = f[(y0i * FW + x1i) * FC_CH + c];
  float v10 = f[(y1i * FW + x0i) * FC_CH + c], v11 = f[(y1i * FW + x1i) * FC_CH + c];
  return v00 * (1 - ly) * (1 - lx) + v01 * (1 - ly) * lx +
         v10 * ly * (1 - lx) + v11 * ly * lx;
}

// 7x7 ROIAlign -> bf16 A matrix [256 rows pad][12544], k = c*49 + oy*7 + ox
__global__ void roi7_kernel(const float* __restrict__ feat,
                            const float* __restrict__ props,
                            __bf16* __restrict__ A1) {
  int n = blockIdx.x;                    // 0..255 (rows >=200 zero-padded)
  __shared__ float sx1, sy1, sbw, sbh;
  bool valid = n < NPROP;
  if (threadIdx.x == 0 && valid) {
    float x1 = props[n * 4 + 0] * SCALE, y1 = props[n * 4 + 1] * SCALE;
    float x2 = props[n * 4 + 2] * SCALE, y2 = props[n * 4 + 3] * SCALE;
    sx1 = x1; sy1 = y1;
    sbw = fmaxf(x2 - x1, 1.0f) / 7.0f;
    sbh = fmaxf(y2 - y1, 1.0f) / 7.0f;
  }
  __syncthreads();
  for (int it = 0; it < 49; ++it) {
    int e = it * 256 + threadIdx.x;      // 0..12543
    float val = 0.0f;
    if (valid) {
      int c = e / 49, p2 = e % 49;
      int oy = p2 / 7, ox = p2 % 7;
      const float* fc = feat + (long)c * (FH * FW);
      float s = 0.0f;
#pragma unroll
      for (int ry = 0; ry < 2; ++ry)
#pragma unroll
        for (int rx = 0; rx < 2; ++rx) {
          float Y = sy1 + ((float)(2 * oy + ry) + 0.5f) * 0.5f * sbh;
          float X = sx1 + ((float)(2 * ox + rx) + 0.5f) * 0.5f * sbw;
          s += bilin_chw(fc, Y, X);
        }
      val = s * 0.25f;
    }
    A1[(long)n * K_FC1 + e] = f2b(val);
  }
}

// 14x14 ROIAlign -> bf16 NHWC [(n*196+p)][256] (A operand layout for deconv)
__global__ void roi14_kernel(const float* __restrict__ fhwc,
                             const float* __restrict__ boxes,
                             __bf16* __restrict__ X1) {
  int n = blockIdx.x, c = threadIdx.x;
  float x1 = boxes[n * 4 + 0] * SCALE, y1 = boxes[n * 4 + 1] * SCALE;
  float x2 = boxes[n * 4 + 2] * SCALE, y2 = boxes[n * 4 + 3] * SCALE;
  float bw = fmaxf(x2 - x1, 1.0f) / 14.0f;
  float bh = fmaxf(y2 - y1, 1.0f) / 14.0f;
  for (int p = 0; p < 196; ++p) {
    int oy = p / 14, ox = p % 14;
    float s = 0.0f;
#pragma unroll
    for (int ry = 0; ry < 2; ++ry)
#pragma unroll
      for (int rx = 0; rx < 2; ++rx) {
        float Y = y1 + ((float)(2 * oy + ry) + 0.5f) * 0.5f * bh;
        float X = x1 + ((float)(2 * ox + rx) + 0.5f) * 0.5f * bw;
        s += bilin_hwc(fhwc, Y, X, c);
      }
    X1[((long)n * 196 + p) * FC_CH + c] = f2b(s * 0.25f);
  }
}

// ------------------------------ GEMM kernels -------------------------------

// C = act(A @ Bt^T + bias), bf16 out (FC layers)
__global__ void gemm_bias_act_bf16(const __bf16* __restrict__ A,
                                   const __bf16* __restrict__ Bt,
                                   const float* __restrict__ bias,
                                   __bf16* __restrict__ C,
                                   int lda, int ldb, int ldc, int K,
                                   int Nreal, int relu) {
  int lane = threadIdx.x & 31, wave = threadIdx.x >> 5;
  int row0 = blockIdx.y * 128 + wave * 16;
  int col0 = blockIdx.x * 64;
  v8f acc[4] = {};
  gemm_tile(A, Bt, lda, ldb, K, row0 + (lane & 15), true, col0, lane, acc);
  int rbase = row0 + 8 * (lane >> 4);
#pragma unroll
  for (int j = 0; j < 4; ++j) {
    int c = col0 + j * 16 + (lane & 15);
    float bv = (c < Nreal) ? bias[c] : 0.0f;
#pragma unroll
    for (int v = 0; v < 8; ++v) {
      float x = acc[j][v] + bv;
      if (relu) x = fmaxf(x, 0.0f);
      C[(long)(rbase + v) * ldc + c] = f2b(x);
    }
  }
}

// C = A @ Bt^T + bias, fp32 out (box / cls heads)
__global__ void gemm_bias_f32(const __bf16* __restrict__ A,
                              const __bf16* __restrict__ Bt,
                              const float* __restrict__ bias,
                              float* __restrict__ C,
                              int lda, int ldb, int ldc, int K, int Nreal) {
  int lane = threadIdx.x & 31, wave = threadIdx.x >> 5;
  int row0 = blockIdx.y * 128 + wave * 16;
  int col0 = blockIdx.x * 64;
  v8f acc[4] = {};
  gemm_tile(A, Bt, lda, ldb, K, row0 + (lane & 15), true, col0, lane, acc);
  int rbase = row0 + 8 * (lane >> 4);
#pragma unroll
  for (int j = 0; j < 4; ++j) {
    int c = col0 + j * 16 + (lane & 15);
    float bv = (c < Nreal) ? bias[c] : 0.0f;
#pragma unroll
    for (int v = 0; v < 8; ++v)
      C[(long)(rbase + v) * ldc + c] = acc[j][v] + bv;
  }
}

// One 2x2-deconv sub-position as GEMM: Y[n,2h+a,2w+b,o] = relu(X[p,:]@W_ab + bias)
__global__ void deconv_gemm(const __bf16* __restrict__ X,
                            const __bf16* __restrict__ Wab,
                            const float* __restrict__ bias,
                            __bf16* __restrict__ Y,
                            int P, int Hin, int Win, int a, int b) {
  int lane = threadIdx.x & 31, wave = threadIdx.x >> 5;
  int row0 = blockIdx.y * 128 + wave * 16;
  int col0 = blockIdx.x * 64;
  int row = row0 + (lane & 15);
  v8f acc[4] = {};
  gemm_tile(X, Wab, FC_CH, FC_CH, FC_CH, row, row < P, col0, lane, acc);
  int rbase = row0 + 8 * (lane >> 4);
  int Wout = 2 * Win;
#pragma unroll
  for (int j = 0; j < 4; ++j) {
    int o = col0 + j * 16 + (lane & 15);
    float bv = bias[o];
#pragma unroll
    for (int v = 0; v < 8; ++v) {
      int p = rbase + v;
      if (p < P) {
        int n = p / (Hin * Win);
        int r = p - n * Hin * Win;
        int h = r / Win, w = r - h * Win;
        float x = fmaxf(acc[j][v] + bv, 0.0f);
        Y[(((long)n * (2 * Hin) + 2 * h + a) * Wout + (2 * w + b)) * FC_CH + o] =
            f2b(x);
      }
    }
  }
}

// mask_logits[n,o,p] = (wm[o,:] @ Y2[n,p,:] + bm[o]) * kf[n]
__global__ void mask_gemm(const __bf16* __restrict__ WM,
                          const __bf16* __restrict__ Y2,
                          const float* __restrict__ bm,
                          const float* __restrict__ kf,
                          float* __restrict__ out) {
  int n = blockIdx.z;
  const __bf16* Bt = Y2 + (long)n * MASK_P * FC_CH;
  int lane = threadIdx.x & 31, wave = threadIdx.x >> 5;
  int row0 = wave * 16;                 // M = 128 (81 padded)
  int col0 = blockIdx.x * 64;           // 49 blocks -> 3136 positions
  v8f acc[4] = {};
  gemm_tile(WM, Bt, FC_CH, FC_CH, FC_CH, row0 + (lane & 15), true, col0, lane, acc);
  float k = kf[n];
  int rbase = row0 + 8 * (lane >> 4);
#pragma unroll
  for (int j = 0; j < 4; ++j) {
    int p = col0 + j * 16 + (lane & 15);
#pragma unroll
    for (int v = 0; v < 8; ++v) {
      int o = rbase + v;
      if (o < NUMC)
        out[((long)n * NUMC + o) * MASK_P + p] = (acc[j][v] + bm[o]) * k;
    }
  }
}

// --------------------------- post-processing -------------------------------

// softmax + argmax + delta decode + clip; 1 block x 256
__global__ void head_post(const float* __restrict__ props,
                          const float* __restrict__ deltas,  // ld 384
                          const float* __restrict__ logits,  // ld 128
                          float* __restrict__ scores,
                          float* __restrict__ boxes,
                          float* __restrict__ bscore) {
  int t = threadIdx.x;
  if (t >= NPROP) return;
  const float* lg = logits + t * 128;
  float m = -1e30f;
  for (int i = 0; i < NUMC; ++i) m = fmaxf(m, lg[i]);
  float s = 0.0f;
  for (int i = 0; i < NUMC; ++i) s += expf(lg[i] - m);
  float inv = 1.0f / s;
  float best = -1.0f; int bi = 0;
  for (int i = 0; i < NUMC; ++i) {
    float p = expf(lg[i] - m) * inv;
    scores[t * NUMC + i] = p;
    if (p > best) { best = p; bi = i; }
  }
  float x1 = props[t * 4 + 0], y1 = props[t * 4 + 1];
  float x2 = props[t * 4 + 2], y2 = props[t * 4 + 3];
  float w = x2 - x1, h = y2 - y1;
  float cx = x1 + 0.5f * w, cy = y1 + 0.5f * h;
  const float* d = deltas + t * 384 + bi * 4;
  float px = d[0] * w + cx, py = d[1] * h + cy;
  float pw = expf(d[2]) * w, ph = expf(d[3]) * h;
  boxes[t * 4 + 0] = fminf(fmaxf(px - 0.5f * pw, 0.0f), IMGSZ);
  boxes[t * 4 + 1] = fminf(fmaxf(py - 0.5f * ph, 0.0f), IMGSZ);
  boxes[t * 4 + 2] = fminf(fmaxf(px + 0.5f * pw, 0.0f), IMGSZ);
  boxes[t * 4 + 3] = fminf(fmaxf(py + 0.5f * ph, 0.0f), IMGSZ);
  bscore[t] = best;
}

// greedy NMS over 200 boxes, 1 block
__global__ void nms_kernel(const float* __restrict__ boxes,
                           const float* __restrict__ bscore,
                           float* __restrict__ kf) {
  __shared__ float bx[NPROP * 4];
  __shared__ float sc[NPROP];
  __shared__ int ord[NPROP];
  __shared__ int keep[NPROP];
  __shared__ int sup;
  int t = threadIdx.x;
  for (int i = t; i < NPROP; i += 256) {
    sc[i] = bscore[i];
    keep[i] = 1;
    for (int j = 0; j < 4; ++j) bx[i * 4 + j] = boxes[i * 4 + j];
  }
  __syncthreads();
  for (int i = t; i < NPROP; i += 256) {       // stable rank of -score
    int r = 0; float si = sc[i];
    for (int j = 0; j < NPROP; ++j) {
      float sj = sc[j];
      if (sj > si || (sj == si && j < i)) ++r;
    }
    ord[r] = i;
  }
  __syncthreads();
  for (int i = 0; i < NPROP; ++i) {
    if (t == 0) sup = 0;
    __syncthreads();
    int bi = ord[i];
    float ax1 = bx[bi*4], ay1 = bx[bi*4+1], ax2 = bx[bi*4+2], ay2 = bx[bi*4+3];
    float aa = fmaxf(ax2 - ax1, 0.0f) * fmaxf(ay2 - ay1, 0.0f);
    for (int jj = t; jj < i; jj += 256) {
      int bj = ord[jj];
      if (keep[bj]) {
        float bx1 = bx[bj*4], by1 = bx[bj*4+1], bx2 = bx[bj*4+2], by2 = bx[bj*4+3];
        float ab = fmaxf(bx2 - bx1, 0.0f) * fmaxf(by2 - by1, 0.0f);
        float ix = fmaxf(fminf(ax2, bx2) - fmaxf(ax1, bx1), 0.0f);
        float iy = fmaxf(fminf(ay2, by2) - fmaxf(ay1, by1), 0.0f);
        float inter = ix * iy;
        float iou = inter / (aa + ab - inter + 1e-9f);
        if (iou > 0.5f) atomicOr(&sup, 1);
      }
    }
    __syncthreads();
    if (t == 0) keep[bi] = (sup == 0);
    __syncthreads();
  }
  for (int i = t; i < NPROP; i += 256) kf[i] = keep[i] ? 1.0f : 0.0f;
}

__global__ void finalize_kernel(const float* __restrict__ boxes,
                                const float* __restrict__ scores,
                                const float* __restrict__ kf,
                                float* __restrict__ out) {
  int idx = blockIdx.x * 256 + threadIdx.x;
  if (idx < NPROP * 4) out[idx] = boxes[idx] * kf[idx >> 2];
  if (idx < NPROP * NUMC) out[NPROP * 4 + idx] = scores[idx] * kf[idx / NUMC];
}

// ------------------------------ launcher -----------------------------------

extern "C" void kernel_launch(void* const* d_in, const int* in_sizes, int n_in,
                              void* d_out, int out_size, void* d_ws, size_t ws_size,
                              hipStream_t stream) {
  const float* feature = (const float*)d_in[0];
  const float* props   = (const float*)d_in[1];
  const float* w1  = (const float*)d_in[2];  const float* b1  = (const float*)d_in[3];
  const float* w2  = (const float*)d_in[4];  const float* b2  = (const float*)d_in[5];
  const float* wbx = (const float*)d_in[6];  const float* bbx = (const float*)d_in[7];
  const float* wcl = (const float*)d_in[8];  const float* bcl = (const float*)d_in[9];
  const float* wt1 = (const float*)d_in[10]; const float* bt1 = (const float*)d_in[11];
  const float* wt2 = (const float*)d_in[12]; const float* bt2 = (const float*)d_in[13];
  const float* wm  = (const float*)d_in[14]; const float* bm  = (const float*)d_in[15];
  float* out = (float*)d_out;

  // workspace bump allocator (256B aligned)
  char* base = (char*)d_ws; size_t off = 0;
  auto alloc = [&](size_t bytes) -> void* {
    void* r = base + off;
    off = (off + bytes + 255) & ~(size_t)255;
    return r;
  };
  float*  fhwc   = (float*) alloc((size_t)FH * FW * FC_CH * 4);        // 67 MB
  __bf16* A1     = (__bf16*)alloc((size_t)256 * K_FC1 * 2);            // 6.4 MB
  __bf16* w1T    = (__bf16*)alloc((size_t)N_FC * K_FC1 * 2);           // 25.7 MB
  __bf16* w2T    = (__bf16*)alloc((size_t)N_FC * N_FC * 2);            // 2 MB
  __bf16* wbxT   = (__bf16*)alloc((size_t)384 * N_FC * 2);
  __bf16* wclT   = (__bf16*)alloc((size_t)128 * N_FC * 2);
  __bf16* h1     = (__bf16*)alloc((size_t)256 * N_FC * 2);
  __bf16* h2     = (__bf16*)alloc((size_t)256 * N_FC * 2);
  float*  deltas = (float*) alloc((size_t)256 * 384 * 4);
  float*  logitb = (float*) alloc((size_t)256 * 128 * 4);
  float*  scores = (float*) alloc((size_t)NPROP * NUMC * 4);
  float*  boxesf = (float*) alloc((size_t)NPROP * 4 * 4);
  float*  bscore = (float*) alloc((size_t)NPROP * 4);
  float*  kf     = (float*) alloc((size_t)NPROP * 4);
  __bf16* bt1c   = (__bf16*)alloc((size_t)4 * 256 * 256 * 2);
  __bf16* bt2c   = (__bf16*)alloc((size_t)4 * 256 * 256 * 2);
  __bf16* wmb    = (__bf16*)alloc((size_t)128 * 256 * 2);
  __bf16* X1     = (__bf16*)alloc((size_t)P_DC1 * FC_CH * 2);          // 20 MB
  __bf16* Y1     = (__bf16*)alloc((size_t)P_DC2 * FC_CH * 2);          // 80 MB
  __bf16* Y2     = (__bf16*)alloc((size_t)NPROP * MASK_P * FC_CH * 2); // 321 MB
  (void)ws_size; (void)in_sizes; (void)n_in; (void)out_size;

  // ---- one-time (per launch) layout conversions ----
  feat_to_hwc<<<dim3(4096, 16), dim3(16, 16), 0, stream>>>(feature, fhwc);
  {
    long t1 = (long)N_FC * K_FC1;
    w_transpose_bf16<<<(int)((t1 + 255) / 256), 256, 0, stream>>>(w1, w1T, K_FC1, N_FC, t1);
    long t2 = (long)N_FC * N_FC;
    w_transpose_bf16<<<(int)((t2 + 255) / 256), 256, 0, stream>>>(w2, w2T, N_FC, N_FC, t2);
    long t3 = (long)384 * N_FC;
    w_transpose_bf16<<<(int)((t3 + 255) / 256), 256, 0, stream>>>(wbx, wbxT, N_FC, NUMC * 4, t3);
    long t4 = (long)128 * N_FC;
    w_transpose_bf16<<<(int)((t4 + 255) / 256), 256, 0, stream>>>(wcl, wclT, N_FC, NUMC, t4);
  }
  wt_convert<<<1024, 256, 0, stream>>>(wt1, bt1c);
  wt_convert<<<1024, 256, 0, stream>>>(wt2, bt2c);
  wm_convert<<<128, 256, 0, stream>>>(wm, wmb);

  // ---- detection branch ----
  roi7_kernel<<<256, 256, 0, stream>>>(feature, props, A1);
  gemm_bias_act_bf16<<<dim3(16, 2), 256, 0, stream>>>(A1, w1T, b1, h1,
      K_FC1, K_FC1, N_FC, K_FC1, N_FC, 1);
  gemm_bias_act_bf16<<<dim3(16, 2), 256, 0, stream>>>(h1, w2T, b2, h2,
      N_FC, N_FC, N_FC, N_FC, N_FC, 1);
  gemm_bias_f32<<<dim3(6, 2), 256, 0, stream>>>(h2, wbxT, bbx, deltas,
      N_FC, N_FC, 384, N_FC, NUMC * 4);
  gemm_bias_f32<<<dim3(2, 2), 256, 0, stream>>>(h2, wclT, bcl, logitb,
      N_FC, N_FC, 128, N_FC, NUMC);
  head_post<<<1, 256, 0, stream>>>(props, deltas, logitb, scores, boxesf, bscore);
  nms_kernel<<<1, 256, 0, stream>>>(boxesf, bscore, kf);
  finalize_kernel<<<64, 256, 0, stream>>>(boxesf, scores, kf, out);

  // ---- mask branch ----
  roi14_kernel<<<NPROP, 256, 0, stream>>>(fhwc, boxesf, X1);
  for (int ab = 0; ab < 4; ++ab)
    deconv_gemm<<<dim3(4, (P_DC1 + 127) / 128), 256, 0, stream>>>(
        X1, bt1c + (size_t)ab * 65536, bt1, Y1, P_DC1, 14, 14, ab >> 1, ab & 1);
  for (int ab = 0; ab < 4; ++ab)
    deconv_gemm<<<dim3(4, P_DC2 / 128), 256, 0, stream>>>(
        Y1, bt2c + (size_t)ab * 65536, bt2, Y2, P_DC2, 28, 28, ab >> 1, ab & 1);
  mask_gemm<<<dim3(49, 1, NPROP), 256, 0, stream>>>(
      wmb, Y2, bm, kf, out + NPROP * 4 + NPROP * NUMC);
}
